// GNN_46471546143165
// MI455X (gfx1250) — compile-verified, matrix operand order checked by reading, not run on hardware
//
#include <hip/hip_runtime.h>
#include <hip/hip_bf16.h>

#define N_NODES 100000
#define HID 128
#define N_CLASSES 16
#define N_GRAPHS 128

typedef __attribute__((ext_vector_type(16))) __bf16 v16bf;
typedef __attribute__((ext_vector_type(8)))  __bf16 v8bf;
typedef __attribute__((ext_vector_type(4)))  __bf16 v4bf;
typedef __attribute__((ext_vector_type(8)))  float  v8f;

// ---------------------------------------------------------------------------
// V_WMMA_F32_16X16X32_BF16 fragment layout (per ISA 16-bit A 16x32 table):
//   lanes 0-15  : row M = lane,    elems 0-7 -> K=k0..k0+7,   8-15 -> K=k0+16..k0+23
//   lanes 16-31 : row M = lane-16, elems 0-7 -> K=k0+8..k0+15, 8-15 -> K=k0+24..k0+31
// Caller passes p already offset by (row stride) + k0 + 8*sub; chunk2 at +16.
// ---------------------------------------------------------------------------

// fragment from LDS bf16 tile: two aligned 16-byte ds loads
__device__ __forceinline__ v16bf lds_frag(const __bf16* p) {
    v8bf lo = *(const v8bf*)(p);
    v8bf hi = *(const v8bf*)(p + 16);
    return __builtin_shufflevector(lo, hi, 0, 1, 2, 3, 4, 5, 6, 7,
                                           8, 9, 10, 11, 12, 13, 14, 15);
}

// fragment from global fp32 (weights): four aligned b128 loads + pack to bf16
__device__ __forceinline__ v16bf glob_frag(const float* __restrict__ p) {
    const float4 a0 = *(const float4*)(p + 0);
    const float4 a1 = *(const float4*)(p + 4);
    const float4 b0 = *(const float4*)(p + 16);
    const float4 b1 = *(const float4*)(p + 20);
    v16bf f;
    f[0]  = (__bf16)a0.x; f[1]  = (__bf16)a0.y; f[2]  = (__bf16)a0.z; f[3]  = (__bf16)a0.w;
    f[4]  = (__bf16)a1.x; f[5]  = (__bf16)a1.y; f[6]  = (__bf16)a1.z; f[7]  = (__bf16)a1.w;
    f[8]  = (__bf16)b0.x; f[9]  = (__bf16)b0.y; f[10] = (__bf16)b0.z; f[11] = (__bf16)b0.w;
    f[12] = (__bf16)b1.x; f[13] = (__bf16)b1.y; f[14] = (__bf16)b1.z; f[15] = (__bf16)b1.w;
    return f;
}

// cooperative stage of a 16 x K fp32 row-block into padded bf16 LDS tile.
// Coalesced b128 global loads; rows clamped (no predicated loads), zeroed by select.
template <int K>
__device__ __forceinline__ void stage_tile(const float* __restrict__ g, int rowBase,
                                           int N, __bf16* __restrict__ lds) {
    constexpr int STRIDE = K + 8;            // bf16 elems; row start stays 16B aligned
#pragma unroll
    for (int it = 0; it < K / 64; ++it) {    // 16*K/4 float4s / 256 threads
        int e4  = threadIdx.x + it * 256;
        int r   = e4 / (K / 4);
        int cc  = e4 % (K / 4);
        int row = rowBase + r;
        int crow = row < N ? row : N - 1;
        float4 v = *(const float4*)(g + (size_t)crow * K + cc * 4);
        if (row >= N) v = make_float4(0.f, 0.f, 0.f, 0.f);
        v4bf pk = { (__bf16)v.x, (__bf16)v.y, (__bf16)v.z, (__bf16)v.w };
        *(v4bf*)(lds + r * STRIDE + cc * 4) = pk;
    }
}

// ---------------------------------------------------------------------------
__global__ void zero_kernel(float4* __restrict__ p, long n4) {
    long i = (long)blockIdx.x * blockDim.x + threadIdx.x;
    long stride = (long)gridDim.x * blockDim.x;
    float4 z = {0.f, 0.f, 0.f, 0.f};
    for (; i < n4; i += stride) p[i] = z;
}

// edge scatter-add: agg[dst[e]] += h[src[e]]   (float4 per thread)
__global__ void scatter_kernel(const float* __restrict__ h,
                               const int* __restrict__ src,
                               const int* __restrict__ dst,
                               float* __restrict__ agg,
                               int F, int shift, long total) {
    long t = (long)blockIdx.x * blockDim.x + threadIdx.x;
    if (t >= total) return;
    long e  = t >> shift;
    int  f4 = (int)(t & ((1 << shift) - 1));
    int  s  = src[e];
    int  d  = dst[e];
    const float4 v = *(const float4*)(h + (size_t)s * F + f4 * 4);
    float* o = agg + (size_t)d * F + f4 * 4;
    atomicAdd(o + 0, v.x);
    atomicAdd(o + 1, v.y);
    atomicAdd(o + 2, v.z);
    atomicAdd(o + 3, v.w);
}

// ---------------------------------------------------------------------------
// Fused GraphConv dense part:
//   out[N x 128] = agg @ Wrel^T + X @ Wroot^T + bias  (+ optional ReLU)
// 256 threads = 8 waves; each wave one 16x16 WMMA tile; A/X tiles staged in LDS.
// ---------------------------------------------------------------------------
template <int K, bool RELU>
__global__ void gemm_dual_kernel(const float* __restrict__ A,
                                 const float* __restrict__ X,
                                 const float* __restrict__ Wrel,
                                 const float* __restrict__ Wroot,
                                 const float* __restrict__ bias,
                                 float* __restrict__ out, int N) {
    constexpr int STRIDE = K + 8;
    __shared__ __bf16 ldsA[16 * STRIDE];
    __shared__ __bf16 ldsX[16 * STRIDE];

    const int rowBase = blockIdx.x * 16;
    stage_tile<K>(A, rowBase, N, ldsA);
    stage_tile<K>(X, rowBase, N, ldsX);
    __syncthreads();

    const int wave = threadIdx.x >> 5;
    const int lane = threadIdx.x & 31;
    const int sub  = lane >> 4;
    const int idx  = lane & 15;
    const int col  = wave * 16 + idx;        // weight row == output column

    const __bf16* aT = ldsA + idx * STRIDE + 8 * sub;
    const __bf16* xT = ldsX + idx * STRIDE + 8 * sub;
    const size_t wOff = (size_t)col * K + 8 * sub;

    v8f c = {};
#pragma unroll
    for (int k0 = 0; k0 < K; k0 += 32) {
        v16bf a = lds_frag(aT + k0);
        v16bf b = glob_frag(Wrel + wOff + k0);
        c = __builtin_amdgcn_wmma_f32_16x16x32_bf16(false, a, false, b,
                                                    (short)0, c, false, false);
    }
#pragma unroll
    for (int k0 = 0; k0 < K; k0 += 32) {
        v16bf a = lds_frag(xT + k0);
        v16bf b = glob_frag(Wroot + wOff + k0);
        c = __builtin_amdgcn_wmma_f32_16x16x32_bf16(false, a, false, b,
                                                    (short)0, c, false, false);
    }

    const float bv = bias[col];
#pragma unroll
    for (int r = 0; r < 8; ++r) {
        int orow = rowBase + r + 8 * sub;    // C/D layout: M = r + 8*sub, N = idx
        if (orow < N) {
            float v = c[r] + bv;
            if (RELU) v = fmaxf(v, 0.0f);
            out[(size_t)orow * HID + col] = v;
        }
    }
}

// ---------------------------------------------------------------------------
// mean-pool pieces
// ---------------------------------------------------------------------------
__global__ void pool_sum_kernel(const float* __restrict__ h,
                                const int* __restrict__ batch,
                                float* __restrict__ sums, long total) {
    long t = (long)blockIdx.x * blockDim.x + threadIdx.x;
    if (t >= total) return;
    long n  = t >> 5;                        // 32 float4 chunks per node
    int  f4 = (int)(t & 31);
    int  g  = batch[n];
    const float4 v = *(const float4*)(h + (size_t)n * HID + f4 * 4);
    float* o = sums + (size_t)g * HID + f4 * 4;
    atomicAdd(o + 0, v.x);
    atomicAdd(o + 1, v.y);
    atomicAdd(o + 2, v.z);
    atomicAdd(o + 3, v.w);
}

__global__ void count_kernel(const int* __restrict__ batch,
                             float* __restrict__ counts, int N) {
    int n = blockIdx.x * blockDim.x + threadIdx.x;
    if (n < N) atomicAdd(&counts[batch[n]], 1.0f);
}

__global__ void div_kernel(float* __restrict__ pooled,
                           const float* __restrict__ counts) {
    int i = blockIdx.x * blockDim.x + threadIdx.x;
    if (i < N_GRAPHS * HID) {
        int g = i >> 7;
        pooled[i] = pooled[i] / fmaxf(counts[g], 1.0f);
    }
}

// ---------------------------------------------------------------------------
// classifier: out[128 x 16] = pooled @ Wlin^T + blin ; 1 block, 8 waves
// ---------------------------------------------------------------------------
__global__ void final_linear_kernel(const float* __restrict__ pooled,
                                    const float* __restrict__ Wlin,
                                    const float* __restrict__ blin,
                                    float* __restrict__ out) {
    const int wave = threadIdx.x >> 5;
    const int lane = threadIdx.x & 31;
    const int sub  = lane >> 4;
    const int idx  = lane & 15;
    const int rowBase = wave * 16;
    const int row = rowBase + idx;

    v8f c = {};
    const size_t aOff = (size_t)row * HID + 8 * sub;
    const size_t wOff = (size_t)idx * HID + 8 * sub;   // Wlin row = class = idx
#pragma unroll
    for (int k0 = 0; k0 < HID; k0 += 32) {
        v16bf a = glob_frag(pooled + aOff + k0);
        v16bf b = glob_frag(Wlin + wOff + k0);
        c = __builtin_amdgcn_wmma_f32_16x16x32_bf16(false, a, false, b,
                                                    (short)0, c, false, false);
    }
    const float bv = blin[idx];
#pragma unroll
    for (int r = 0; r < 8; ++r) {
        int orow = rowBase + r + 8 * sub;
        out[(size_t)orow * N_CLASSES + idx] = c[r] + bv;
    }
}

// ---------------------------------------------------------------------------
extern "C" void kernel_launch(void* const* d_in, const int* in_sizes, int n_in,
                              void* d_out, int out_size, void* d_ws, size_t ws_size,
                              hipStream_t stream) {
    const float* x      = (const float*)d_in[0];
    const int*   eidx   = (const int*)d_in[1];
    const int*   batch  = (const int*)d_in[2];
    const float* W1_rel = (const float*)d_in[3];
    const float* b1     = (const float*)d_in[4];
    const float* W1_rt  = (const float*)d_in[5];
    const float* W2_rel = (const float*)d_in[6];
    const float* b2     = (const float*)d_in[7];
    const float* W2_rt  = (const float*)d_in[8];
    const float* W3_rel = (const float*)d_in[9];
    const float* b3     = (const float*)d_in[10];
    const float* W3_rt  = (const float*)d_in[11];
    const float* W_lin  = (const float*)d_in[12];
    const float* b_lin  = (const float*)d_in[13];
    float* out = (float*)d_out;

    const long nEdges = in_sizes[1] / 2;
    const int* src = eidx;
    const int* dst = eidx + nEdges;
    const int  N   = N_NODES;

    // workspace layout
    float* bufAgg = (float*)d_ws;                       // N*128
    float* bufH1  = bufAgg + (size_t)N * HID;           // N*128
    float* bufH2  = bufH1  + (size_t)N * HID;           // N*128
    float* pooled = bufH2  + (size_t)N * HID;           // 128*128
    float* counts = pooled + (size_t)N_GRAPHS * HID;    // 128

    const int TB = 256;
    dim3 gemmGrid((N + 15) / 16);

    // ---- layer 1 (F_IN = 64) ----
    {
        long n4 = (long)N * 64 / 4;
        zero_kernel<<<2048, TB, 0, stream>>>((float4*)bufAgg, n4);
        long total = nEdges * 16;               // F/4 = 16, shift = 4
        scatter_kernel<<<(total + TB - 1) / TB, TB, 0, stream>>>(
            x, src, dst, bufAgg, 64, 4, total);
        gemm_dual_kernel<64, true><<<gemmGrid, TB, 0, stream>>>(
            bufAgg, x, W1_rel, W1_rt, b1, bufH1, N);
    }
    // ---- layer 2 (HID = 128) ----
    {
        long n4 = (long)N * HID / 4;
        zero_kernel<<<2048, TB, 0, stream>>>((float4*)bufAgg, n4);
        long total = nEdges * 32;               // F/4 = 32, shift = 5
        scatter_kernel<<<(total + TB - 1) / TB, TB, 0, stream>>>(
            bufH1, src, dst, bufAgg, HID, 5, total);
        gemm_dual_kernel<128, true><<<gemmGrid, TB, 0, stream>>>(
            bufAgg, bufH1, W2_rel, W2_rt, b2, bufH2, N);
    }
    // ---- layer 3 (HID = 128, no ReLU) ----
    {
        long n4 = (long)N * HID / 4;
        zero_kernel<<<2048, TB, 0, stream>>>((float4*)bufAgg, n4);
        long total = nEdges * 32;
        scatter_kernel<<<(total + TB - 1) / TB, TB, 0, stream>>>(
            bufH2, src, dst, bufAgg, HID, 5, total);
        gemm_dual_kernel<128, false><<<gemmGrid, TB, 0, stream>>>(
            bufAgg, bufH2, W3_rel, W3_rt, b3, bufH1, N);   // h3 -> bufH1
    }
    // ---- global mean pool ----
    {
        long nz4 = (N_GRAPHS * HID + HID) / 4;   // sums + counts
        zero_kernel<<<64, TB, 0, stream>>>((float4*)pooled, nz4);
        long total = (long)N * 32;
        pool_sum_kernel<<<(total + TB - 1) / TB, TB, 0, stream>>>(
            bufH1, batch, pooled, total);
        count_kernel<<<(N + TB - 1) / TB, TB, 0, stream>>>(batch, counts, N);
        div_kernel<<<(N_GRAPHS * HID + TB - 1) / TB, TB, 0, stream>>>(pooled, counts);
    }
    // ---- classifier ----
    final_linear_kernel<<<1, TB, 0, stream>>>(pooled, W_lin, b_lin, out);
}
